// SelfSupPredictor_42949673119
// MI455X (gfx1250) — matrix operand order repaired and checked by reading, not compiled, not run on hardware
//
#include <hip/hip_runtime.h>
#include <math.h>

#define N_NODES 50000
#define N_EDGES 200000
#define S_STEPS 32
#define K_TRIPLES 16384
#define D_MAX 4
#define F_IN 16
#define F_DIM 128

typedef __bf16 bf16;
typedef __attribute__((ext_vector_type(16))) __bf16 v16bf;
typedef __attribute__((ext_vector_type(8)))  float  v8f;
typedef __attribute__((ext_vector_type(4)))  unsigned u32x4;
typedef __attribute__((ext_vector_type(4)))  int      v4i;

union BU { u32x4 u[2]; v16bf v; };

__device__ inline v8f v8f_zero() {
  v8f z;
#pragma unroll
  for (int i = 0; i < 8; i++) z[i] = 0.f;
  return z;
}

__device__ inline void add4(float4& a, const float4 b) {
  a.x += b.x; a.y += b.y; a.z += b.z; a.w += b.w;
}

__device__ inline v16bf to_bf16x16(const float4 a0, const float4 a1,
                                   const float4 a2, const float4 a3) {
  v16bf a;
  a[0] = (bf16)a0.x; a[1] = (bf16)a0.y; a[2]  = (bf16)a0.z; a[3]  = (bf16)a0.w;
  a[4] = (bf16)a1.x; a[5] = (bf16)a1.y; a[6]  = (bf16)a1.z; a[7]  = (bf16)a1.w;
  a[8] = (bf16)a2.x; a[9] = (bf16)a2.y; a[10] = (bf16)a2.z; a[11] = (bf16)a2.w;
  a[12] = (bf16)a3.x; a[13] = (bf16)a3.y; a[14] = (bf16)a3.z; a[15] = (bf16)a3.w;
  return a;
}

// Block-cooperative staging of `bytes` (multiple of 16) from global into LDS.
// Uses the CDNA5 async global->LDS path when available (ASYNCcnt), else a
// plain load/store copy. Ends with a workgroup barrier.
__device__ inline void stage_to_lds(const char* g, char* l, int bytes) {
  const int tid = threadIdx.x;
#if __has_builtin(__builtin_amdgcn_global_load_async_to_lds_b128)
  for (int o = tid * 16; o < bytes; o += 256 * 16) {
    __builtin_amdgcn_global_load_async_to_lds_b128(
        (__attribute__((address_space(1))) v4i*)(g + o),
        (__attribute__((address_space(3))) v4i*)(l + o), 0, 0);
  }
#if __has_builtin(__builtin_amdgcn_s_wait_asynccnt)
  __builtin_amdgcn_s_wait_asynccnt(0);
#else
  asm volatile("s_wait_asynccnt 0x0" ::: "memory");
#endif
#else
  for (int o = tid * 16; o < bytes; o += 256 * 16) {
    *(u32x4*)(l + o) = *(const u32x4*)(g + o);
  }
#endif
  __syncthreads();
}

// ---------------------------------------------------------------- setup

__global__ void k_init(float* out, unsigned* cnt) {
  int i = blockIdx.x * blockDim.x + threadIdx.x;
  if (i < N_NODES) { out[i] = 0.f; cnt[i] = 0u; }
}

__global__ void k_zero_u32(unsigned* p) {
  int i = blockIdx.x * blockDim.x + threadIdx.x;
  if (i < N_NODES) p[i] = 0u;
}

__global__ void k_count(const long long* dst_e, unsigned* cnt) {
  int e = blockIdx.x * blockDim.x + threadIdx.x;
  if (e < N_EDGES) atomicAdd(&cnt[(int)dst_e[e]], 1u);
}

// single-block exclusive scan of cnt -> row_ptr, plus deg_inv
__global__ void k_scan(const unsigned* cnt, unsigned* row_ptr, float* deg_inv) {
  __shared__ unsigned sh[1024];
  const int t = threadIdx.x;
  unsigned running = 0;
  for (int base = 0; base < N_NODES; base += 1024) {
    int i = base + t;
    unsigned v = (i < N_NODES) ? cnt[i] : 0u;
    sh[t] = v;
    __syncthreads();
    for (int ofs = 1; ofs < 1024; ofs <<= 1) {
      unsigned add = (t >= ofs) ? sh[t - ofs] : 0u;
      __syncthreads();
      sh[t] += add;
      __syncthreads();
    }
    unsigned incl = sh[t];
    if (i < N_NODES) {
      row_ptr[i] = running + incl - v;
      deg_inv[i] = 1.0f / fmaxf((float)v, 1.0f);
    }
    unsigned total = sh[1023];
    __syncthreads();
    running += total;
  }
  if (t == 0) row_ptr[N_NODES] = running;
}

__global__ void k_fill(const long long* src_e, const long long* dst_e,
                       const unsigned* row_ptr, unsigned* cursor, unsigned* col_src) {
  int e = blockIdx.x * blockDim.x + threadIdx.x;
  if (e < N_EDGES) {
    int d = (int)dst_e[e];
    unsigned p = atomicAdd(&cursor[d], 1u);
    col_src[row_ptr[d] + p] = (unsigned)src_e[e];
  }
}

// pack W [Kdim x 128] (row-major) into WMMA B-operand tiles, two interleaved
// 16B chunks per lane (lane stride 16B -> conflict-free LDS b128 reads):
//   chunk layout: packed[((tile*2 + c)*32 + lane)*8 + j], elem i = c*8 + j
//   value: bf16(W[kt*32 + (lane/16)*8 + i + (i/8)*8][nt*16 + lane%16]), tile = kt*8+nt
__global__ void k_pack(const float* W, int Kdim, int KT, bf16* packed) {
  int t = blockIdx.x * blockDim.x + threadIdx.x;
  int total = KT * 8 * 32;
  if (t >= total) return;
  int lane = t & 31;
  int tile = t >> 5;
  int kt = tile >> 3;
  int nt = tile & 7;
  int n = nt * 16 + (lane & 15);
  int kbase = kt * 32 + (lane >> 4) * 8;
#pragma unroll
  for (int i = 0; i < 16; i++) {
    int k = kbase + i + (i >> 3) * 8;
    float v = (k < Kdim) ? W[(size_t)k * F_DIM + n] : 0.f;
    int c = i >> 3, j = i & 7;
    packed[((size_t)(tile * 2 + c) * 32 + lane) * 8 + j] = (bf16)v;
  }
}

// ---------------------------------------------------------------- aggregation

__global__ void k_agg16(const float* x, const unsigned* row_ptr, const unsigned* col_src,
                        const float* deg_inv, float* aggbuf) {
  int t = blockIdx.x * blockDim.x + threadIdx.x;
  if (t >= N_NODES * F_IN) return;
  int n = t >> 4;
  int f = t & 15;
  unsigned b = row_ptr[n], e = row_ptr[n + 1];
  float acc = 0.f;
  for (unsigned j = b; j < e; j++) acc += x[(size_t)col_src[j] * F_IN + f];
  aggbuf[t] = x[t] + acc * deg_inv[n];
}

__global__ __launch_bounds__(256) void k_agg128(const float* __restrict__ feats,
                                                const unsigned* __restrict__ row_ptr,
                                                const unsigned* __restrict__ col_src,
                                                const float* __restrict__ deg_inv,
                                                float* __restrict__ aggbuf) {
  int wv = (blockIdx.x * blockDim.x + threadIdx.x) >> 5;
  int lane = threadIdx.x & 31;
  if (wv >= N_NODES) return;
  int n = wv;
  float4 acc = make_float4(0.f, 0.f, 0.f, 0.f);
  unsigned b = row_ptr[n], e = row_ptr[n + 1];
  for (unsigned j = b; j < e; j++) {
    unsigned s = col_src[j];
    float4 v = ((const float4*)(feats + (size_t)s * F_DIM))[lane];
    add4(acc, v);
  }
  float di = deg_inv[n];
  float4 sf = ((const float4*)(feats + (size_t)n * F_DIM))[lane];
  float4 o;
  o.x = sf.x + acc.x * di;
  o.y = sf.y + acc.y * di;
  o.z = sf.z + acc.z * di;
  o.w = sf.w + acc.w * di;
  ((float4*)(aggbuf + (size_t)n * F_DIM))[lane] = o;
}

// ---------------------------------------------------------------- encoder GEMM
// Out[row,:] = relu(A[row,0:KDIM] @ W + bias). W pre-packed bf16, async-staged
// into LDS once per block; one wave per 16-row tile; fully unrolled K loop.
template <int KDIM>
__global__ __launch_bounds__(256) void k_gemm(const float* __restrict__ A,
                                              const u32x4* __restrict__ Bp,
                                              const float* __restrict__ bias,
                                              float* __restrict__ Out, int nTiles) {
  constexpr int KT = (KDIM + 31) / 32;
  extern __shared__ u32x4 smem_u4[];
  stage_to_lds((const char*)Bp, (char*)smem_u4, KT * 8 * 2 * 32 * 16);

  int wv = (blockIdx.x * 256 + threadIdx.x) >> 5;
  int lane = threadIdx.x & 31;
  if (wv >= nTiles) return;
  int half = lane >> 4;
  int m = lane & 15;
  const float* arow = A + (size_t)(wv * 16 + m) * KDIM;

  v8f acc[8];
#pragma unroll
  for (int i = 0; i < 8; i++) acc[i] = v8f_zero();

#pragma unroll
  for (int kt = 0; kt < KT; kt++) {
    int k0 = kt * 32 + half * 8;   // A elems 0..7  -> K = k0..k0+7
    int k1 = k0 + 16;              // A elems 8..15 -> K = k1..k1+7
    float4 a0 = make_float4(0, 0, 0, 0), a1 = a0, a2 = a0, a3 = a0;
    if (k0 + 7 < KDIM) {
      const float4* p = (const float4*)(arow + k0);
      a0 = p[0]; a1 = p[1];
    }
    if (k1 + 7 < KDIM) {
      const float4* q = (const float4*)(arow + k1);
      a2 = q[0]; a3 = q[1];
    }
    v16bf a = to_bf16x16(a0, a1, a2, a3);

    BU bu[8];
#pragma unroll
    for (int nt = 0; nt < 8; nt++) {
      int tile = kt * 8 + nt;
      bu[nt].u[0] = smem_u4[(tile * 2 + 0) * 32 + lane];
      bu[nt].u[1] = smem_u4[(tile * 2 + 1) * 32 + lane];
    }
#pragma unroll
    for (int nt = 0; nt < 8; nt++) {
      acc[nt] = __builtin_amdgcn_wmma_f32_16x16x32_bf16(
          false, a, false, bu[nt].v, (short)0, acc[nt], false, false);
    }
  }

#pragma unroll
  for (int nt = 0; nt < 8; nt++) {
    int col = nt * 16 + m;
    float bv = bias[col];
#pragma unroll
    for (int r = 0; r < 8; r++) {
      int row = wv * 16 + half * 8 + r;      // C: M = r + 8*half
      float v = acc[nt][r] + bv;
      Out[(size_t)row * F_DIM + col] = fmaxf(v, 0.f);
    }
  }
}

// ---------------------------------------------------------------- fused decoder
// One wave per 16 triples: gather [pf|sf|df] row fragments directly, WMMA vs
// packed W_dec1 (12 K-tiles, register-batched B loads), ReLU, dot with W_dec2
// via shfl_xor reduce, log-sigmoid, masked atomic add into out[place].
__global__ __launch_bounds__(256) void k_decoder(
    const float* __restrict__ feats, const long long* __restrict__ place_idx,
    const long long* __restrict__ src_idx, const long long* __restrict__ dst_idx,
    const unsigned char* __restrict__ triple_mask, const unsigned char* __restrict__ dst_mask,
    const u32x4* __restrict__ Bd1, const float* __restrict__ b_dec1,
    const float* __restrict__ W_dec2, const float* __restrict__ b_dec2,
    int s, float* __restrict__ out) {
  int wv = (blockIdx.x * 256 + threadIdx.x) >> 5;
  int lane = threadIdx.x & 31;
  if (wv >= K_TRIPLES / 16) return;
  int half = lane >> 4;
  int m = lane & 15;
  int t = wv * 16 + m;                       // triple index of this A row
  size_t tb = (size_t)s * K_TRIPLES + t;

  const float* pf = feats + (size_t)place_idx[tb] * F_DIM;
  const float* sf = feats + (size_t)src_idx[tb] * F_DIM;
  const float* df[4];
  int dm[4];
#pragma unroll
  for (int j = 0; j < 4; j++) {
    dm[j] = dst_mask[tb * D_MAX + j];
    df[j] = feats + (size_t)dst_idx[tb * D_MAX + j] * F_DIM;
  }

  v8f acc[8];
#pragma unroll
  for (int i = 0; i < 8; i++) acc[i] = v8f_zero();

#pragma unroll
  for (int kt = 0; kt < 12; kt++) {
    const int seg = kt >> 2;                  // 0=pf 1=sf 2=df (compile-time)
    int inner = (kt & 3) * 32 + half * 8;     // run0 at inner, run1 at inner+16
    float4 r0a, r0b, r1a, r1b;
    if (seg == 0) {
      const float4* p = (const float4*)(pf + inner);
      r0a = p[0]; r0b = p[1];
      const float4* q = (const float4*)(pf + inner + 16);
      r1a = q[0]; r1b = q[1];
    } else if (seg == 1) {
      const float4* p = (const float4*)(sf + inner);
      r0a = p[0]; r0b = p[1];
      const float4* q = (const float4*)(sf + inner + 16);
      r1a = q[0]; r1b = q[1];
    } else {
      r0a = make_float4(0, 0, 0, 0); r0b = r0a; r1a = r0a; r1b = r0a;
#pragma unroll
      for (int j = 0; j < 4; j++) {
        if (dm[j]) {
          const float4* p = (const float4*)(df[j] + inner);
          add4(r0a, p[0]); add4(r0b, p[1]);
          const float4* q = (const float4*)(df[j] + inner + 16);
          add4(r1a, q[0]); add4(r1b, q[1]);
        }
      }
    }
    v16bf a = to_bf16x16(r0a, r0b, r1a, r1b);

    BU bu[8];
#pragma unroll
    for (int nt = 0; nt < 8; nt++) {
      int tile = kt * 8 + nt;
      bu[nt].u[0] = Bd1[(size_t)(tile * 2 + 0) * 32 + lane];
      bu[nt].u[1] = Bd1[(size_t)(tile * 2 + 1) * 32 + lane];
    }
#pragma unroll
    for (int nt = 0; nt < 8; nt++) {
      acc[nt] = __builtin_amdgcn_wmma_f32_16x16x32_bf16(
          false, a, false, bu[nt].v, (short)0, acc[nt], false, false);
    }
  }

  // second decoder layer: h @ W_dec2 (128 -> 1), reduced across the 16-lane half
  float w2[8], b1v[8];
#pragma unroll
  for (int nt = 0; nt < 8; nt++) {
    w2[nt] = W_dec2[nt * 16 + m];
    b1v[nt] = b_dec1[nt * 16 + m];
  }
  float tot[8];
#pragma unroll
  for (int r = 0; r < 8; r++) {
    float p = 0.f;
#pragma unroll
    for (int nt = 0; nt < 8; nt++) {
      float h = fmaxf(acc[nt][r] + b1v[nt], 0.f);
      p += h * w2[nt];
    }
    for (int ofs = 1; ofs < 16; ofs <<= 1) p += __shfl_xor(p, ofs, 32);
    tot[r] = p;
  }
  if (m == 0) {
    float b2 = b_dec2[0];
#pragma unroll
    for (int r = 0; r < 8; r++) {
      int trip = wv * 16 + half * 8 + r;     // C row -> triple
      size_t ti = (size_t)s * K_TRIPLES + trip;
      if (triple_mask[ti]) {
        float xv = tot[r] + b2;
        float lp = fminf(xv, 0.f) - log1pf(expf(-fabsf(xv)));
        atomicAdd(&out[(size_t)place_idx[ti]], lp);
      }
    }
  }
}

// ---------------------------------------------------------------- launch

extern "C" void kernel_launch(void* const* d_in, const int* in_sizes, int n_in,
                              void* d_out, int out_size, void* d_ws, size_t ws_size,
                              hipStream_t stream) {
  const float* x = (const float*)d_in[0];
  const long long* edge_index = (const long long*)d_in[1];
  const long long* src_e = edge_index;
  const long long* dst_e = edge_index + N_EDGES;
  const long long* place_idx = (const long long*)d_in[2];
  const long long* src_idx = (const long long*)d_in[3];
  const long long* dst_idx = (const long long*)d_in[4];
  const unsigned char* triple_mask = (const unsigned char*)d_in[5];
  const unsigned char* dst_mask = (const unsigned char*)d_in[6];
  const float* W_enc1 = (const float*)d_in[7];
  const float* b_enc1 = (const float*)d_in[8];
  const float* W_enc2 = (const float*)d_in[9];
  const float* b_enc2 = (const float*)d_in[10];
  const float* W_dec1 = (const float*)d_in[11];
  const float* b_dec1 = (const float*)d_in[12];
  const float* W_dec2 = (const float*)d_in[13];
  const float* b_dec2 = (const float*)d_in[14];
  float* out = (float*)d_out;

  char* ws = (char*)d_ws;
  size_t off = 0;
  auto take = [&](size_t bytes) -> void* {
    void* p = (void*)(ws + off);
    off += (bytes + 255) & ~(size_t)255;
    return p;
  };

  unsigned* cnt      = (unsigned*)take(sizeof(unsigned) * N_NODES);
  unsigned* row_ptr  = (unsigned*)take(sizeof(unsigned) * (N_NODES + 1));
  unsigned* col_src  = (unsigned*)take(sizeof(unsigned) * N_EDGES);
  float*    deg_inv  = (float*)take(sizeof(float) * N_NODES);
  float*    featsA   = (float*)take(sizeof(float) * (size_t)N_NODES * F_DIM);
  float*    featsB   = (float*)take(sizeof(float) * (size_t)N_NODES * F_DIM);
  float*    aggbuf   = (float*)take(sizeof(float) * (size_t)N_NODES * F_DIM);
  bf16*     packW1   = (bf16*)take(sizeof(bf16) * (size_t)1 * 8 * 32 * 16);
  bf16*     packW2   = (bf16*)take(sizeof(bf16) * (size_t)4 * 8 * 32 * 16);
  bf16*     packWd1  = (bf16*)take(sizeof(bf16) * (size_t)12 * 8 * 32 * 16);
  (void)in_sizes; (void)n_in; (void)out_size; (void)ws_size;

  const int nTiles = N_NODES / 16;  // 3125 exact
  dim3 blk(256);

  k_init<<<(N_NODES + 255) / 256, blk, 0, stream>>>(out, cnt);
  k_count<<<(N_EDGES + 255) / 256, blk, 0, stream>>>(dst_e, cnt);
  k_scan<<<1, 1024, 0, stream>>>(cnt, row_ptr, deg_inv);
  k_zero_u32<<<(N_NODES + 255) / 256, blk, 0, stream>>>(cnt);
  k_fill<<<(N_EDGES + 255) / 256, blk, 0, stream>>>(src_e, dst_e, row_ptr, cnt, col_src);

  k_pack<<<(1 * 8 * 32 + 255) / 256, blk, 0, stream>>>(W_enc1, F_IN, 1, packW1);
  k_pack<<<(4 * 8 * 32 + 255) / 256, blk, 0, stream>>>(W_enc2, F_DIM, 4, packW2);
  k_pack<<<(12 * 8 * 32 + 255) / 256, blk, 0, stream>>>(W_dec1, 3 * F_DIM, 12, packWd1);

  // first encoder: feats0 = relu((x + msgs*deg_inv) @ W_enc1 + b_enc1)
  k_agg16<<<(N_NODES * F_IN + 255) / 256, blk, 0, stream>>>(x, row_ptr, col_src, deg_inv, aggbuf);
  k_gemm<F_IN><<<(nTiles + 7) / 8, blk, 1 * 8 * 2 * 32 * 16, stream>>>(
      aggbuf, (const u32x4*)packW1, b_enc1, featsA, nTiles);

  float* cur = featsA;
  float* nxt = featsB;
  for (int s = 0; s < S_STEPS; s++) {
    k_decoder<<<(K_TRIPLES / 16 + 7) / 8, blk, 0, stream>>>(
        cur, place_idx, src_idx, dst_idx, triple_mask, dst_mask,
        (const u32x4*)packWd1, b_dec1, W_dec2, b_dec2, s, out);
    if (s + 1 < S_STEPS) {  // last step's feats are unused by the reference
      k_agg128<<<(N_NODES + 7) / 8, blk, 0, stream>>>(cur, row_ptr, col_src, deg_inv, aggbuf);
      k_gemm<F_DIM><<<(nTiles + 7) / 8, blk, 4 * 8 * 2 * 32 * 16, stream>>>(
          aggbuf, (const u32x4*)packW2, b_enc2, nxt, nTiles);
      float* tmp = cur; cur = nxt; nxt = tmp;
    }
  }
}
